// CRISPRO_MambaX_4827543241039
// MI455X (gfx1250) — matrix operand order repaired and checked by reading, not compiled, not run on hardware
//
#include <hip/hip_runtime.h>
#include <hip/hip_bf16.h>

// ---------------- model constants (from reference) ----------------
constexpr int B_   = 2;
constexpr int L_   = 4096;
constexpr int DM_  = 512;
constexpr int NL_  = 6;
constexpr int DI_  = 1024;
constexpr int HD_  = 64;
constexpr int NH_  = 16;
constexpr int NS_  = 128;
constexpr int CD_  = DI_ + 2 * NS_;            // 1280
constexpr int DIP_ = 2 * DI_ + 2 * NS_ + NH_;  // 2320
constexpr int KW_  = 4;
constexpr int CH_  = 256;
constexpr int NC_  = L_ / CH_;                 // 16
constexpr int NT_  = 5;
constexpr long long BL_ = (long long)B_ * L_;  // 8192

constexpr int LDS_STRIDE = 40;  // ushorts per row (80 B, keeps 16-B alignment)

// ---------------- WMMA plumbing (gfx1250, wave32) ----------------
typedef __attribute__((ext_vector_type(16))) __bf16 v16bf;
typedef __attribute__((ext_vector_type(8)))  __bf16 v8bf;
typedef __attribute__((ext_vector_type(8)))  float  v8f;
typedef __attribute__((ext_vector_type(4)))  unsigned int u32x4;

__device__ __forceinline__ unsigned short f32_to_bf16_bits(float f) {
  unsigned u = __builtin_bit_cast(unsigned, f);
  unsigned r = (u + 0x7FFFu + ((u >> 16) & 1u)) >> 16;   // round-to-nearest-even
  return (unsigned short)r;
}
__device__ __forceinline__ float bf16_bits_to_f32(unsigned short u) {
  return __builtin_bit_cast(float, (unsigned)u << 16);
}
__device__ __forceinline__ v8f zero8() {
  v8f z = {0.f, 0.f, 0.f, 0.f, 0.f, 0.f, 0.f, 0.f};
  return z;
}

// A fragment: row-major LDS row (k contiguous). Lane needs K = hi*8..+7 and
// 16+hi*8..+7  -> two 16-byte LDS reads (ds_load_b128).
__device__ __forceinline__ v16bf fragA(const unsigned short* row, int hi) {
  v8bf lo = *(const v8bf*)(row + hi * 8);
  v8bf hs = *(const v8bf*)(row + 16 + hi * 8);
  return __builtin_shufflevector(lo, hs, 0, 1, 2, 3, 4, 5, 6, 7,
                                 8, 9, 10, 11, 12, 13, 14, 15);
}
// B fragment: LDS holds B transposed (one column per row, k contiguous).
__device__ __forceinline__ v16bf fragB(const unsigned short* row, int hi) {
  v8bf lo = *(const v8bf*)(row + hi * 8);
  v8bf hs = *(const v8bf*)(row + 16 + hi * 8);
  return __builtin_shufflevector(lo, hs, 0, 1, 2, 3, 4, 5, 6, 7,
                                 8, 9, 10, 11, 12, 13, 14, 15);
}
__device__ __forceinline__ v8f wmma_bf16(v16bf a, v16bf b, v8f acc) {
  return __builtin_amdgcn_wmma_f32_16x16x32_bf16(false, a, false, b, (short)0,
                                                 acc, false, false);
}

// ---------------- weight convert+transpose: dst[n][k] = bf16(src[k][n]) ----
__global__ void k_wtrans(const float* __restrict__ src, unsigned short* __restrict__ dst,
                         int K, int N) {
  long long i = blockIdx.x * 256ll + threadIdx.x;
  if (i >= (long long)K * N) return;
  int k = (int)(i / N), n = (int)(i % N);
  dst[(long long)n * K + k] = f32_to_bf16_bits(src[i]);
}

// ---------------- bf16 GEMM: C = A(MxK) * B(KxN), B given transposed -------
// A: bf16 row-major M x K (lda).  Bt: bf16 row-major N x K (ldbt).
// Block = 256 threads = 8 waves -> 64x64 C tile; each wave owns two 16x16
// tiles sharing one A fragment. Staging = one b128 global load + one b128
// LDS store per thread per operand per K-step (no conversion in the loop).
__global__ void k_gemm_bf16(const unsigned short* __restrict__ Ap, int lda, long long sA,
                            const unsigned short* __restrict__ Bt, int ldbt, long long sBT,
                            float* __restrict__ Cp, int ldc, long long sC,
                            int M, int N, int K, const float* __restrict__ bias) {
  __shared__ alignas(16) unsigned short As[64 * LDS_STRIDE];  // [m][k]
  __shared__ alignas(16) unsigned short Bs[64 * LDS_STRIDE];  // [n][k]
  long long bz = blockIdx.z;
  Ap += bz * sA; Bt += bz * sBT; Cp += bz * sC;
  int m0g = blockIdx.y * 64;
  int n0g = blockIdx.x * 64;
  int tid = threadIdx.x;
  int wave = tid >> 5;
  int wm = (wave & 3) * 16;    // 0..48
  int wn = (wave >> 2) * 16;   // 0 / 16  (second tile at wn+32)
  int lane = tid & 31, lid = lane & 15, hi = lane >> 4;
  int rowT = tid >> 2;         // 0..63
  int col8 = (tid & 3) * 8;    // 0,8,16,24
  int mG = m0g + rowT, nG = n0g + rowT;
  const u32x4 zz = {0u, 0u, 0u, 0u};
  v8f acc0 = zero8(), acc1 = zero8();
  for (int k0 = 0; k0 < K; k0 += 32) {
    u32x4 va = zz, vb = zz;
    if (mG < M) va = *(const u32x4*)(Ap + (long long)mG * lda + k0 + col8);
    if (nG < N) vb = *(const u32x4*)(Bt + (long long)nG * ldbt + k0 + col8);
    *(u32x4*)(As + rowT * LDS_STRIDE + col8) = va;
    *(u32x4*)(Bs + rowT * LDS_STRIDE + col8) = vb;
    if (k0 + 32 < K) {  // global_prefetch_b8 for next K-tile
      if (mG < M) __builtin_prefetch(Ap + (long long)mG * lda + k0 + 32 + col8, 0, 1);
      if (nG < N) __builtin_prefetch(Bt + (long long)nG * ldbt + k0 + 32 + col8, 0, 1);
    }
    __syncthreads();
    v16bf a  = fragA(As + (wm + lid) * LDS_STRIDE, hi);
    v16bf b0 = fragB(Bs + (wn + lid) * LDS_STRIDE, hi);
    v16bf b1 = fragB(Bs + (wn + 32 + lid) * LDS_STRIDE, hi);
    acc0 = wmma_bf16(a, b0, acc0);
    acc1 = wmma_bf16(a, b1, acc1);
    __syncthreads();
  }
#pragma unroll
  for (int r = 0; r < 8; ++r) {
    int m = m0g + wm + hi * 8 + r;
    if (m < M) {
      int n = n0g + wn + lid;
      if (n < N) Cp[(long long)m * ldc + n] = acc0[r] + (bias ? bias[n] : 0.f);
      int n2 = n + 32;
      if (n2 < N) Cp[(long long)m * ldc + n2] = acc1[r] + (bias ? bias[n2] : 0.f);
    }
  }
}

// ---------------- prologue: embedding + track attention fusion -------------
__global__ void k_embed_fuse(const int* __restrict__ seq,
                             const float* __restrict__ epi,
                             const float* __restrict__ emb,
                             const float* __restrict__ Wt,
                             const float* __restrict__ bt,
                             const float* __restrict__ Wa,
                             const float* __restrict__ ba,
                             float* __restrict__ xbuf,
                             unsigned short* __restrict__ xfb) {
  long long tok = blockIdx.x;  // b*L + l
  int lane = threadIdx.x;      // 32 lanes, 16 channels each
  float x[16];
  int v = seq[tok];
  const float* e = emb + (long long)v * DM_;
#pragma unroll
  for (int i = 0; i < 16; ++i) x[i] = e[lane * 16 + i];
  float logit[NT_];
#pragma unroll
  for (int t = 0; t < NT_; ++t) {
    float p = 0.f;
#pragma unroll
    for (int i = 0; i < 16; ++i) p += x[i] * Wa[(lane * 16 + i) * NT_ + t];
#pragma unroll
    for (int o = 16; o > 0; o >>= 1) p += __shfl_xor(p, o, 32);
    logit[t] = p + ba[t];
  }
  float mx = logit[0];
#pragma unroll
  for (int t = 1; t < NT_; ++t) mx = fmaxf(mx, logit[t]);
  float se = 0.f, al[NT_];
#pragma unroll
  for (int t = 0; t < NT_; ++t) { al[t] = __expf(logit[t] - mx); se += al[t]; }
  float inv = 1.f / se;
  float ep[NT_];
#pragma unroll
  for (int t = 0; t < NT_; ++t) ep[t] = epi[tok * NT_ + t];
  float* xo = xbuf + tok * DM_;
  unsigned short* xfo = xfb + tok * 2 * DM_;
#pragma unroll
  for (int i = 0; i < 16; ++i) {
    int d = lane * 16 + i;
    float f = 0.f;
#pragma unroll
    for (int t = 0; t < NT_; ++t)
      f += (al[t] * inv) * (ep[t] * Wt[t * DM_ + d] + bt[t * DM_ + d]);
    xo[d] = x[i];
    xfo[d] = f32_to_bf16_bits(x[i]);
    xfo[DM_ + d] = f32_to_bf16_bits(f);
  }
}

__global__ void k_gate_combine(float* __restrict__ x,
                               const float* __restrict__ glin,
                               const unsigned short* __restrict__ xfb) {
  long long i = blockIdx.x * 256ll + threadIdx.x;
  long long n = BL_ * DM_;
  if (i >= n) return;
  long long tok = i / DM_;
  int d = (int)(i % DM_);
  float g = 1.f / (1.f + __expf(-glin[i]));
  x[i] += g * bf16_bits_to_f32(xfb[tok * 2 * DM_ + DM_ + d]);
}

// ---------------- layernorm over DM=512; writes f32 + bf16 copies ----------
__global__ void k_layernorm(const float* __restrict__ src, float* __restrict__ dst,
                            unsigned short* __restrict__ dstb,
                            const float* __restrict__ w, const float* __restrict__ b) {
  __shared__ float red[256];
  long long tok = blockIdx.x;
  int tid = threadIdx.x;
  const float* s = src + tok * DM_;
  float v0 = s[tid], v1 = s[tid + 256];
  red[tid] = v0 + v1;
  __syncthreads();
  for (int o = 128; o > 0; o >>= 1) { if (tid < o) red[tid] += red[tid + o]; __syncthreads(); }
  float mean = red[0] / DM_;
  __syncthreads();
  float d0 = v0 - mean, d1 = v1 - mean;
  red[tid] = d0 * d0 + d1 * d1;
  __syncthreads();
  for (int o = 128; o > 0; o >>= 1) { if (tid < o) red[tid] += red[tid + o]; __syncthreads(); }
  float rstd = rsqrtf(red[0] / DM_ + 1e-5f);
  float o0 = d0 * rstd * w[tid] + b[tid];
  float o1 = d1 * rstd * w[tid + 256] + b[tid + 256];
  float* ot = dst + tok * DM_;
  ot[tid] = o0;
  ot[tid + 256] = o1;
  unsigned short* ob = dstb + tok * DM_;
  ob[tid] = f32_to_bf16_bits(o0);
  ob[tid + 256] = f32_to_bf16_bits(o1);
}

__global__ void k_flip_bf(const unsigned short* __restrict__ src,
                          unsigned short* __restrict__ dst) {
  long long i = blockIdx.x * 256ll + threadIdx.x;
  long long n = BL_ * DM_;
  if (i >= n) return;
  long long bl = i / DM_;
  int d = (int)(i % DM_);
  int b = (int)(bl / L_), l = (int)(bl % L_);
  dst[i] = src[((long long)b * L_ + (L_ - 1 - l)) * DM_ + d];
}

// ---------------- causal depthwise conv (KW=4) + SiLU ----------------------
// Writes f32 xc (all CD channels) + bf16 copy of the B/C channels (2*NS).
__global__ void k_conv_silu(const float* __restrict__ zx,
                            const float* __restrict__ cw,
                            const float* __restrict__ cb,
                            float* __restrict__ xc,
                            unsigned short* __restrict__ bcb) {
  long long i = blockIdx.x * 256ll + threadIdx.x;
  long long n = BL_ * CD_;
  if (i >= n) return;
  int ch = (int)(i % CD_);
  long long bl = i / CD_;
  int b = (int)(bl / L_), l = (int)(bl % L_);
  float acc = cb[ch];
#pragma unroll
  for (int t = 0; t < KW_; ++t) {
    int ls = l - (KW_ - 1) + t;
    if (ls >= 0)
      acc += zx[((long long)b * L_ + ls) * DIP_ + DI_ + ch] * cw[ch * KW_ + t];
  }
  float sv = acc / (1.f + __expf(-acc));  // silu
  xc[i] = sv;
  if (ch >= DI_) bcb[bl * (2 * NS_) + (ch - DI_)] = f32_to_bf16_bits(sv);
}

// ---------------- dt = softplus(dt_raw + bias); a = -exp(A_log)*dt ---------
__global__ void k_dt_a(const float* __restrict__ zx, const float* __restrict__ db,
                       const float* __restrict__ Al, float* __restrict__ dtb,
                       float* __restrict__ ab) {
  long long i = blockIdx.x * 256ll + threadIdx.x;
  long long n = BL_ * NH_;
  if (i >= n) return;
  int h = (int)(i % NH_);
  long long bl = i / NH_;
  float raw = zx[bl * DIP_ + DI_ + CD_ + h] + db[h];
  float dt = raw > 20.f ? raw : log1pf(__expf(raw));
  dtb[i] = dt;
  ab[i] = -__expf(Al[h]) * dt;
}

__global__ void k_xdt(const float* __restrict__ xc, const float* __restrict__ dtb,
                      unsigned short* __restrict__ xdtb) {
  long long i = blockIdx.x * 256ll + threadIdx.x;
  long long n = BL_ * DI_;
  if (i >= n) return;
  int d = (int)(i % DI_);
  long long bl = i / DI_;
  xdtb[i] = f32_to_bf16_bits(xc[bl * CD_ + d] * dtb[bl * NH_ + (d >> 6)]);  // HD=64
}

// ---------------- per-chunk inclusive cumsum of a (wave32 scan) ------------
__global__ void k_cumsum(const float* __restrict__ ab, float* __restrict__ acs) {
  int z = blockIdx.x;         // (b*NH + h)*NC + c
  int lane = threadIdx.x;
  int c = z % NC_;
  int bh = z / NC_;
  int b = bh / NH_, h = bh % NH_;
  int lbase = c * CH_ + lane * 8;
  float v[8];
  float loc = 0.f;
#pragma unroll
  for (int j = 0; j < 8; ++j) {
    v[j] = ab[((long long)b * L_ + lbase + j) * NH_ + h];
    loc += v[j];
  }
  float s = loc;
  for (int o = 1; o < 32; o <<= 1) {
    float t = __shfl_up(s, o, 32);
    if (lane >= o) s += t;
  }
  float cum = s - loc;
#pragma unroll
  for (int j = 0; j < 8; ++j) {
    cum += v[j];
    acs[((long long)b * NH_ + h) * L_ + lbase + j] = cum;
  }
}

// ---------------- SSD diagonal block: Yd = (mask.decay.S) @ (x*dt) ---------
// z -> (b,c,h); blockIdx.y -> 64-row l-tile; N = HD = 64; K = CH = 256.
__global__ void k_yd(const float* __restrict__ S, const float* __restrict__ acs,
                     const unsigned short* __restrict__ xdtb, float* __restrict__ y) {
  __shared__ alignas(16) unsigned short Ws[64 * LDS_STRIDE];  // [l][s]
  __shared__ alignas(16) unsigned short Xs[64 * LDS_STRIDE];  // [p][s]
  int z = blockIdx.z;
  int h = z % NH_;
  int c = (z / NH_) % NC_;
  int b = z / (NH_ * NC_);
  int l0 = blockIdx.y * 64;
  int cbase = c * CH_;
  const float* acsh = acs + ((long long)b * NH_ + h) * L_ + cbase;
  const float* Sc = S + (long long)(b * NC_ + c) * CH_ * CH_;
  int tid = threadIdx.x;
  int wave = tid >> 5;
  int wm = (wave & 3) * 16, wn = (wave >> 2) * 16;
  int lane = tid & 31, lid = lane & 15, hi = lane >> 4;
  v8f acc0 = zero8(), acc1 = zero8();
  for (int k0 = 0; k0 < CH_; k0 += 32) {
#pragma unroll
    for (int t = 0; t < 8; ++t) {
      int idx = tid + t * 256;
      int r = idx >> 5, cc = idx & 31;
      int l = l0 + r, s = k0 + cc;
      float w = 0.f;
      if (s <= l) w = Sc[l * CH_ + s] * __expf(acsh[l] - acsh[s]);
      Ws[r * LDS_STRIDE + cc] = f32_to_bf16_bits(w);
      int kk = idx >> 6, n = idx & 63;
      Xs[n * LDS_STRIDE + kk] =
          xdtb[(((long long)b * L_ + cbase + k0 + kk) * NH_ + h) * HD_ + n];
    }
    __syncthreads();
    v16bf a  = fragA(Ws + (wm + lid) * LDS_STRIDE, hi);
    v16bf b0 = fragB(Xs + (wn + lid) * LDS_STRIDE, hi);
    v16bf b1 = fragB(Xs + (wn + 32 + lid) * LDS_STRIDE, hi);
    acc0 = wmma_bf16(a, b0, acc0);
    acc1 = wmma_bf16(a, b1, acc1);
    __syncthreads();
  }
#pragma unroll
  for (int r = 0; r < 8; ++r) {
    int l = l0 + wm + hi * 8 + r;
    long long yb = (((long long)b * L_ + cbase + l) * NH_ + h) * HD_;
    y[yb + wn + lid]      = acc0[r];
    y[yb + wn + 32 + lid] = acc1[r];
  }
}

// ---------------- chunk states: st[p,n] = sum_l dec(l)*x[l,p] * B[l,n] -----
// z -> (b,c,h); M=HD=64 (single y tile), N=NS=128 (x=2 tiles), K=CH=256.
__global__ void k_states(const unsigned short* __restrict__ xdtb,
                         const unsigned short* __restrict__ bcb,
                         const float* __restrict__ acs, float* __restrict__ st) {
  __shared__ alignas(16) unsigned short As[64 * LDS_STRIDE];  // [p][l]
  __shared__ alignas(16) unsigned short Bs[64 * LDS_STRIDE];  // [n][l]
  int z = blockIdx.z;
  int h = z % NH_;
  int c = (z / NH_) % NC_;
  int b = z / (NH_ * NC_);
  int n0 = blockIdx.x * 64;
  int cbase = c * CH_;
  const float* acsh = acs + ((long long)b * NH_ + h) * L_ + cbase;
  float aL = acsh[CH_ - 1];
  int tid = threadIdx.x;
  int wave = tid >> 5;
  int wm = (wave & 3) * 16, wn = (wave >> 2) * 16;
  int lane = tid & 31, lid = lane & 15, hi = lane >> 4;
  v8f acc0 = zero8(), acc1 = zero8();
  for (int k0 = 0; k0 < CH_; k0 += 32) {
#pragma unroll
    for (int t = 0; t < 8; ++t) {
      int idx = tid + t * 256;
      int r = idx >> 5, cc = idx & 31;   // A: p x l
      int l = k0 + cc;
      float dec = __expf(aL - acsh[l]);
      float xv = bf16_bits_to_f32(
          xdtb[(((long long)b * L_ + cbase + l) * NH_ + h) * HD_ + r]);
      As[r * LDS_STRIDE + cc] = f32_to_bf16_bits(xv * dec);
      int kk = idx >> 6, n = idx & 63;   // B: l x n (pure bf16 copy)
      Bs[n * LDS_STRIDE + kk] =
          bcb[((long long)b * L_ + cbase + k0 + kk) * (2 * NS_) + n0 + n];
    }
    __syncthreads();
    v16bf a  = fragA(As + (wm + lid) * LDS_STRIDE, hi);
    v16bf b0 = fragB(Bs + (wn + lid) * LDS_STRIDE, hi);
    v16bf b1 = fragB(Bs + (wn + 32 + lid) * LDS_STRIDE, hi);
    acc0 = wmma_bf16(a, b0, acc0);
    acc1 = wmma_bf16(a, b1, acc1);
    __syncthreads();
  }
  long long sb = ((long long)(b * NC_ + c) * NH_ + h) * (HD_ * NS_);
#pragma unroll
  for (int r = 0; r < 8; ++r) {
    int p = wm + hi * 8 + r;
    st[sb + (long long)p * NS_ + n0 + wn + lid]      = acc0[r];
    st[sb + (long long)p * NS_ + n0 + wn + 32 + lid] = acc1[r];
  }
}

// ---------------- inter-chunk recurrence (16 steps, sequential) ------------
__global__ void k_scan(const float* __restrict__ st, const float* __restrict__ acs,
                       float* __restrict__ ent) {
  int bh = blockIdx.x;
  int b = bh / NH_, h = bh % NH_;
  int tid = threadIdx.x;
  float s[32];
#pragma unroll
  for (int j = 0; j < 32; ++j) s[j] = 0.f;
  for (int c = 0; c < NC_; ++c) {
    float cdec = __expf(acs[((long long)b * NH_ + h) * L_ + c * CH_ + CH_ - 1]);
    long long base = ((long long)(b * NC_ + c) * NH_ + h) * (HD_ * NS_);
#pragma unroll
    for (int j = 0; j < 32; ++j) {
      long long idx = base + tid + j * 256;
      ent[idx] = s[j];
      s[j] = cdec * s[j] + st[idx];
    }
  }
}

// ---------------- state output: y += exp(acs[l]) * C @ entT ----------------
// z -> (b,c,h); M=CH=256 (y=4 tiles of 64), N=HD=64, K=NS=128.
__global__ void k_yo(const unsigned short* __restrict__ bcb,
                     const float* __restrict__ ent,
                     const float* __restrict__ acs, float* __restrict__ y) {
  __shared__ alignas(16) unsigned short As[64 * LDS_STRIDE];  // [l][n]
  __shared__ alignas(16) unsigned short Bs[64 * LDS_STRIDE];  // [p][n]
  int z = blockIdx.z;
  int h = z % NH_;
  int c = (z / NH_) % NC_;
  int b = z / (NH_ * NC_);
  int l0 = blockIdx.y * 64;
  int cbase = c * CH_;
  const float* acsh = acs + ((long long)b * NH_ + h) * L_ + cbase;
  long long ebase = ((long long)(b * NC_ + c) * NH_ + h) * (HD_ * NS_);
  int tid = threadIdx.x;
  int wave = tid >> 5;
  int wm = (wave & 3) * 16, wn = (wave >> 2) * 16;
  int lane = tid & 31, lid = lane & 15, hi = lane >> 4;
  v8f acc0 = zero8(), acc1 = zero8();
  for (int k0 = 0; k0 < NS_; k0 += 32) {
#pragma unroll
    for (int t = 0; t < 8; ++t) {
      int idx = tid + t * 256;
      int r = idx >> 5, cc = idx & 31;   // A: l x n_state (bf16 copy of C)
      As[r * LDS_STRIDE + cc] =
          bcb[((long long)b * L_ + cbase + l0 + r) * (2 * NS_) + NS_ + k0 + cc];
      int kk = idx >> 6, n = idx & 63;   // B: n_state x p (ent[p][n] transposed)
      Bs[n * LDS_STRIDE + kk] = f32_to_bf16_bits(
          ent[ebase + (long long)n * NS_ + k0 + kk]);
    }
    __syncthreads();
    v16bf a  = fragA(As + (wm + lid) * LDS_STRIDE, hi);
    v16bf b0 = fragB(Bs + (wn + lid) * LDS_STRIDE, hi);
    v16bf b1 = fragB(Bs + (wn + 32 + lid) * LDS_STRIDE, hi);
    acc0 = wmma_bf16(a, b0, acc0);
    acc1 = wmma_bf16(a, b1, acc1);
    __syncthreads();
  }
#pragma unroll
  for (int r = 0; r < 8; ++r) {
    int l = l0 + wm + hi * 8 + r;
    float sc = __expf(acsh[l]);
    long long yb = (((long long)b * L_ + cbase + l) * NH_ + h) * HD_;
    y[yb + wn + lid]      += sc * acc0[r];
    y[yb + wn + 32 + lid] += sc * acc1[r];
  }
}

// ---------------- (y + D*x) * silu(z), gated RMSNorm over DI ---------------
__global__ void k_gate_rms(const float* __restrict__ y, const float* __restrict__ xc,
                           const float* __restrict__ zx, const float* __restrict__ Dp,
                           const float* __restrict__ nw, unsigned short* __restrict__ gb) {
  __shared__ float red[256];
  long long tok = blockIdx.x;
  int tid = threadIdx.x;
  float v[4];
  float ss = 0.f;
#pragma unroll
  for (int j = 0; j < 4; ++j) {
    int d = tid + j * 256;
    float zz = zx[tok * DIP_ + d];
    float xs = xc[tok * CD_ + d];
    float val = (y[tok * DI_ + d] + Dp[d >> 6] * xs) * (zz / (1.f + __expf(-zz)));
    v[j] = val;
    ss += val * val;
  }
  red[tid] = ss;
  __syncthreads();
  for (int o = 128; o > 0; o >>= 1) { if (tid < o) red[tid] += red[tid + o]; __syncthreads(); }
  float r = rsqrtf(red[0] / DI_ + 1e-5f);
#pragma unroll
  for (int j = 0; j < 4; ++j) {
    int d = tid + j * 256;
    gb[tok * DI_ + d] = f32_to_bf16_bits(v[j] * r * nw[d]);
  }
}

// ---------------- residual combine: x += fwd + flip(rev) -------------------
__global__ void k_combine(float* __restrict__ x, const float* __restrict__ f,
                          const float* __restrict__ r) {
  long long i = blockIdx.x * 256ll + threadIdx.x;
  long long n = BL_ * DM_;
  if (i >= n) return;
  long long bl = i / DM_;
  int d = (int)(i % DM_);
  int b = (int)(bl / L_), l = (int)(bl % L_);
  x[i] += f[i] + r[((long long)b * L_ + (L_ - 1 - l)) * DM_ + d];
}

// ---------------- mean pool over L ----------------------------------------
__global__ void k_mean(const float* __restrict__ xn, float* __restrict__ lat) {
  int i = blockIdx.x * 256 + threadIdx.x;
  if (i >= B_ * DM_) return;
  int b = i / DM_, d = i % DM_;
  float s = 0.f;
  for (int l = 0; l < L_; ++l) s += xn[((long long)b * L_ + l) * DM_ + d];
  lat[i] = s / (float)L_;
}

// ---------------- MLP head: sigmoid(leaky(lat@W1+b1)@W2+b2) ----------------
__global__ void k_head(const float* __restrict__ lat, const float* __restrict__ W1,
                       const float* __restrict__ b1, const float* __restrict__ W2,
                       const float* __restrict__ b2, float* __restrict__ o) {
  __shared__ float red[256];
  int b = blockIdx.x;
  int j = threadIdx.x;
  const float* la = lat + (long long)b * DM_;
  float h = b1[j];
  for (int d = 0; d < DM_; ++d) h += la[d] * W1[d * 256 + j];
  float lr = h > 0.f ? h : 0.1f * h;
  red[j] = lr * W2[j];
  __syncthreads();
  for (int o2 = 128; o2 > 0; o2 >>= 1) { if (j < o2) red[j] += red[j + o2]; __syncthreads(); }
  if (j == 0) {
    float s = red[0] + b2[0];
    o[b] = 1.f / (1.f + __expf(-s));
  }
}

// ======================= host-side orchestration ===========================
extern "C" void kernel_launch(void* const* d_in, const int* in_sizes, int n_in,
                              void* d_out, int out_size, void* d_ws, size_t ws_size,
                              hipStream_t stream) {
  (void)in_sizes; (void)n_in; (void)out_size; (void)ws_size;
  const int*   seq     = (const int*)d_in[0];
  const float* epi     = (const float*)d_in[1];
  const float* emb     = (const float*)d_in[2];
  const float* W_track = (const float*)d_in[3];
  const float* b_track = (const float*)d_in[4];
  const float* W_attn  = (const float*)d_in[5];
  const float* b_attn  = (const float*)d_in[6];
  const float* W_gate  = (const float*)d_in[7];
  const float* b_gate  = (const float*)d_in[8];
  const float* ln_w    = (const float*)d_in[9];
  const float* ln_b    = (const float*)d_in[10];
  const float* W_in    = (const float*)d_in[11];
  const float* conv_w  = (const float*)d_in[12];
  const float* conv_b  = (const float*)d_in[13];
  const float* A_log   = (const float*)d_in[14];
  const float* dt_bias = (const float*)d_in[15];
  const float* Dp      = (const float*)d_in[16];
  const float* norm_w  = (const float*)d_in[17];
  const float* W_out   = (const float*)d_in[18];
  const float* fln_w   = (const float*)d_in[19];
  const float* fln_b   = (const float*)d_in[20];
  const float* on_W1   = (const float*)d_in[21];
  const float* on_b1   = (const float*)d_in[22];
  const float* on_W2   = (const float*)d_in[23];
  const float* on_b2   = (const float*)d_in[24];
  const float* off_W1  = (const float*)d_in[25];
  const float* off_b1  = (const float*)d_in[26];
  const float* off_W2  = (const float*)d_in[27];
  const float* off_b2  = (const float*)d_in[28];
  float* out = (float*)d_out;  // [on(2) | off(2) | lat(2*512)]

  // ---- workspace carve (32-byte aligned blocks, with aliasing) ----
  float* Wk = (float*)d_ws;
  long long off = 0;
  auto alloc = [&](long long nf) { float* p = Wk + off; off += (nf + 7) & ~7ll; return p; };
  auto alloc_us = [&](long long nu) { return (unsigned short*)alloc((nu + 1) / 2); };
  float* xbuf   = alloc(BL_ * DM_);
  float* region = alloc(BL_ * DIP_);           // gate_lin (prologue) / zx (layers)
  float* gate_lin = region;
  float* zx       = region;
  float* xc    = alloc(BL_ * CD_);
  float* dtb   = alloc(BL_ * NH_);
  float* ab    = alloc(BL_ * NH_);
  float* acs   = alloc(BL_ * NH_);
  float* Sbuf  = alloc((long long)B_ * NC_ * CH_ * CH_);
  float* stbuf = alloc((long long)B_ * NC_ * NH_ * HD_ * NS_);
  float* ent   = alloc((long long)B_ * NC_ * NH_ * HD_ * NS_);
  float* ybuf  = alloc(BL_ * DI_);
  float* fbuf  = alloc(BL_ * DM_);
  float* rbuf  = alloc(BL_ * DM_);
  float* xn    = alloc(BL_ * DM_);
  // bf16 operand pools
  unsigned short* wg_t  = alloc_us((long long)2 * DM_ * DM_ * 2);        // [DM][2DM]
  unsigned short* wi_t  = alloc_us((long long)NL_ * 2 * DIP_ * DM_);     // [ld][DIP][DM]
  unsigned short* wo_t  = alloc_us((long long)NL_ * 2 * DM_ * DI_);      // [ld][DM][DI]
  unsigned short* xf_bf = alloc_us(BL_ * 2 * DM_);
  unsigned short* u_bf  = alloc_us(BL_ * DM_);
  unsigned short* ur_bf = alloc_us(BL_ * DM_);
  unsigned short* g_bf  = alloc_us(BL_ * DI_);
  unsigned short* bc_bf = alloc_us(BL_ * 2 * NS_);
  unsigned short* xdt_bf = alloc_us(BL_ * DI_);

  auto gemm = [&](const unsigned short* Ap, int lda, long long sA,
                  const unsigned short* Bt, int ldbt, long long sBT,
                  float* Cp, int ldc, long long sC,
                  int M, int N, int K, int batch, const float* bias) {
    dim3 g((N + 63) / 64, (M + 63) / 64, batch), blk(256);
    k_gemm_bf16<<<g, blk, 0, stream>>>(Ap, lda, sA, Bt, ldbt, sBT, Cp, ldc, sC,
                                       M, N, K, bias);
  };
  auto grid1 = [](long long n) { return dim3((unsigned)((n + 255) / 256)); };

  // ---- weight convert + transpose (bf16, N x K) ----
  k_wtrans<<<grid1((long long)2 * DM_ * DM_), 256, 0, stream>>>(W_gate, wg_t, 2 * DM_, DM_);
  for (int ld = 0; ld < NL_ * 2; ++ld) {
    k_wtrans<<<grid1((long long)DM_ * DIP_), 256, 0, stream>>>(
        W_in + (long long)ld * DM_ * DIP_, wi_t + (long long)ld * DIP_ * DM_, DM_, DIP_);
    k_wtrans<<<grid1((long long)DI_ * DM_), 256, 0, stream>>>(
        W_out + (long long)ld * DI_ * DM_, wo_t + (long long)ld * DM_ * DI_, DI_, DM_);
  }

  // ---- prologue: embed + track fusion + gated residual ----
  k_embed_fuse<<<(unsigned)BL_, 32, 0, stream>>>(seq, epi, emb, W_track, b_track,
                                                 W_attn, b_attn, xbuf, xf_bf);
  gemm(xf_bf, 2 * DM_, 0, wg_t, 2 * DM_, 0, gate_lin, DM_, 0,
       (int)BL_, DM_, 2 * DM_, 1, b_gate);
  k_gate_combine<<<grid1(BL_ * DM_), 256, 0, stream>>>(xbuf, gate_lin, xf_bf);

  // ---- 6 bidirectional Mamba blocks ----
  for (int i = 0; i < NL_; ++i) {
    // f32 LN output is scratch (only bf16 copy is consumed); park it in zx.
    k_layernorm<<<(unsigned)BL_, 256, 0, stream>>>(xbuf, zx, u_bf,
                                                   ln_w + i * DM_, ln_b + i * DM_);
    k_flip_bf<<<grid1(BL_ * DM_), 256, 0, stream>>>(u_bf, ur_bf);
    for (int dir = 0; dir < 2; ++dir) {
      int ld = i * 2 + dir;
      const unsigned short* uin = dir ? ur_bf : u_bf;
      const float* cw = conv_w + (long long)ld * CD_ * KW_;
      const float* cb = conv_b + (long long)ld * CD_;
      const float* Al = A_log + (long long)ld * NH_;
      const float* db = dt_bias + (long long)ld * NH_;
      const float* Dh = Dp + (long long)ld * NH_;
      const float* nw = norm_w + (long long)ld * DI_;
      float* dout = dir ? rbuf : fbuf;

      // in-projection: [8192,512] @ [512,2320]
      gemm(uin, DM_, 0, wi_t + (long long)ld * DIP_ * DM_, DM_, 0,
           zx, DIP_, 0, (int)BL_, DIP_, DM_, 1, nullptr);
      k_conv_silu<<<grid1(BL_ * CD_), 256, 0, stream>>>(zx, cw, cb, xc, bc_bf);
      k_dt_a<<<grid1(BL_ * NH_), 256, 0, stream>>>(zx, db, Al, dtb, ab);
      k_cumsum<<<B_ * NH_ * NC_, 32, 0, stream>>>(ab, acs);
      k_xdt<<<grid1(BL_ * DI_), 256, 0, stream>>>(xc, dtb, xdt_bf);

      // chunk scores S = C @ B^T (activations already in N x K form)
      gemm(bc_bf + NS_, 2 * NS_, (long long)CH_ * 2 * NS_,
           bc_bf,       2 * NS_, (long long)CH_ * 2 * NS_,
           Sbuf, CH_, (long long)CH_ * CH_, CH_, CH_, NS_, B_ * NC_, nullptr);
      k_yd<<<dim3(1, CH_ / 64, B_ * NC_ * NH_), 256, 0, stream>>>(Sbuf, acs, xdt_bf, ybuf);
      k_states<<<dim3(NS_ / 64, 1, B_ * NC_ * NH_), 256, 0, stream>>>(xdt_bf, bc_bf, acs, stbuf);
      k_scan<<<B_ * NH_, 256, 0, stream>>>(stbuf, acs, ent);
      k_yo<<<dim3(1, CH_ / 64, B_ * NC_ * NH_), 256, 0, stream>>>(bc_bf, ent, acs, ybuf);

      k_gate_rms<<<(unsigned)BL_, 256, 0, stream>>>(ybuf, xc, zx, Dh, nw, g_bf);
      // out-projection: [8192,1024] @ [1024,512]
      gemm(g_bf, DI_, 0, wo_t + (long long)ld * DM_ * DI_, DI_, 0,
           dout, DM_, 0, (int)BL_, DM_, DI_, 1, nullptr);
    }
    k_combine<<<grid1(BL_ * DM_), 256, 0, stream>>>(xbuf, fbuf, rbuf);
  }

  // ---- final LN, mean pool, heads ----
  k_layernorm<<<(unsigned)BL_, 256, 0, stream>>>(xbuf, xn, u_bf, fln_w, fln_b);
  k_mean<<<grid1(B_ * DM_), 256, 0, stream>>>(xn, out + 4);
  k_head<<<B_, 256, 0, stream>>>(out + 4, on_W1, on_b1, on_W2, on_b2, out);
  k_head<<<B_, 256, 0, stream>>>(out + 4, off_W1, off_b1, off_W2, off_b2, out + 2);
}
